// Gen_skip_SRNN_DEXAT_mix_84009560310441
// MI455X (gfx1250) — matrix operand order repaired and checked by reading, not compiled, not run on hardware
//
#include <hip/hip_runtime.h>

typedef _Float16 f16x8  __attribute__((ext_vector_type(8)));
typedef _Float16 f16x16 __attribute__((ext_vector_type(16)));
typedef float    f32x8  __attribute__((ext_vector_type(8)));
typedef unsigned int u32x4 __attribute__((ext_vector_type(4)));
typedef int          i32x4 __attribute__((ext_vector_type(4)));
typedef int          i32x8 __attribute__((ext_vector_type(8)));

static constexpr int NBATCH = 64, STEPS = 784, INDIM = 128, HDIM = 512, ODIM = 10;
static constexpr int NWG = 16;     // persistent blocks (trivially co-resident)
static constexpr int TPB = 256;    // 8 wave32 per block -> 128 waves = 128 tiles
static constexpr size_t NH   = (size_t)NBATCH * HDIM;            // 32768
static constexpr size_t XH_E = (size_t)NBATCH * STEPS * INDIM;   // 6422528

// membrane dynamics constants (float32 of exp(-1/tau))
#define ALPHA 0.95122945f
#define RO1   0.96721610f
#define RO2   0.99667221f

// ---- d_out element offsets (tuple return order, flattened) ----
static constexpr size_t O_OUT  = 0;                                // [64,784,10]
static constexpr size_t O_M1   = (size_t)NBATCH * STEPS * ODIM;    // 501760
static constexpr size_t O_M2   = O_M1 + NH;
static constexpr size_t O_M3   = O_M2 + NH;
static constexpr size_t O_S1   = O_M3 + NH;                        // s1f,s2f,s3f contiguous
static constexpr size_t O_FIRE = O_S1 + 3 * NH;
static constexpr size_t O_HS   = O_FIRE + 1;                       // hidden_spike [64,784,512]
static constexpr size_t O_HM   = O_HS + (size_t)NBATCH * STEPS * HDIM;
static constexpr size_t O_HT   = O_HM + (size_t)NBATCH * STEPS * HDIM;

// ---- workspace byte offsets ----
static constexpr size_t W_CTRL = 0;                                 // cnt, gen, fire (u32)
static constexpr size_t W_XH   = 64;                                // x as f16
static constexpr size_t W_WI1  = W_XH  + XH_E * 2;                  // 512x128 f16
static constexpr size_t W_WH1  = W_WI1 + (size_t)HDIM * INDIM * 2;  // 512x512 f16 x5 follow
static constexpr size_t W_WI2  = W_WH1 + (size_t)HDIM * HDIM * 2;
static constexpr size_t W_WH2  = W_WI2 + (size_t)HDIM * HDIM * 2;
static constexpr size_t W_WI3  = W_WH2 + (size_t)HDIM * HDIM * 2;
static constexpr size_t W_WH3  = W_WI3 + (size_t)HDIM * HDIM * 2;
static constexpr size_t W_SB   = W_WH3 + (size_t)HDIM * HDIM * 2;   // 6 * NH f16 (ping-pong spikes)
static constexpr size_t W_BST  = W_SB  + 6 * NH * 2;                // 6 * NH f32 (b11..b32)

// LDS layout (dynamic shared): A1 tile at 0, A2 tile at 64KB
static constexpr unsigned LDS_A1 = 0;
static constexpr unsigned LDS_A2 = 65536;
static constexpr unsigned LDS_BYTES = 131072;

// ---------------- grid-wide barrier (sense-reversing, agent scope) ----------------
__device__ inline void grid_barrier(unsigned* cnt, unsigned* gen) {
  __threadfence();
  __syncthreads();
  if (threadIdx.x == 0) {
    unsigned g = __hip_atomic_load(gen, __ATOMIC_RELAXED, __HIP_MEMORY_SCOPE_AGENT);
    unsigned a = __hip_atomic_fetch_add(cnt, 1u, __ATOMIC_ACQ_REL, __HIP_MEMORY_SCOPE_AGENT);
    if (a == (unsigned)(NWG - 1)) {
      __hip_atomic_store(cnt, 0u, __ATOMIC_RELAXED, __HIP_MEMORY_SCOPE_AGENT);
      __hip_atomic_fetch_add(gen, 1u, __ATOMIC_RELEASE, __HIP_MEMORY_SCOPE_AGENT);
    } else {
      while (__hip_atomic_load(gen, __ATOMIC_ACQUIRE, __HIP_MEMORY_SCOPE_AGENT) == g)
        __builtin_amdgcn_s_sleep(1);
    }
  }
  __syncthreads();
  __threadfence();
}

// ---------------- TDM: stage a 2D f16 tile (rows x cols, row stride in elems) into LDS ----------------
// D# bitfields per CDNA5 ISA 8.3/8.4: group0 {count=1, lds_addr, global_addr, type=2},
// group1 {data_size=2B, tensor_dim0/1, tile_dim0/1, tensor_dim0_stride}.
__device__ inline void tdm_stage(unsigned lds_off, const void* gptr,
                                 unsigned cols, unsigned rows, unsigned long row_stride_e) {
  unsigned long ga = (unsigned long)gptr;
  u32x4 g0;
  g0.x = 1u;                                              // count=1, user mode
  g0.y = lds_off;                                         // lds_addr (bytes)
  g0.z = (unsigned)(ga & 0xffffffffu);                    // global_addr[31:0]
  g0.w = (unsigned)((ga >> 32) & 0x01ffffffu) | (2u << 30);  // global_addr[56:32], type=2
  i32x8 g1;
  g1[0] = 1 << 16;                                        // data_size=1 (2 bytes), wg_mask=0
  g1[1] = (int)((cols & 0xffffu) << 16);                  // tensor_dim0[15:0]
  g1[2] = (int)(((cols >> 16) & 0xffffu) | ((rows & 0xffffu) << 16));   // td0[31:16], td1[15:0]
  g1[3] = (int)(((rows >> 16) & 0xffffu) | ((cols & 0xffffu) << 16));   // td1[31:16], tile_dim0
  g1[4] = (int)(rows & 0xffffu);                          // tile_dim1 (tile_dim2=0)
  g1[5] = (int)(row_stride_e & 0xffffffffu);              // tensor_dim0_stride[31:0]
  g1[6] = (int)((row_stride_e >> 32) & 0xffffu);          // stride[47:32]; dim1_stride unused
  g1[7] = 0;
  i32x4 z4 = {0, 0, 0, 0};
  i32x8 z8 = {0, 0, 0, 0, 0, 0, 0, 0};
  __builtin_amdgcn_tensor_load_to_lds(g0, g1, z4, z4, z8, 0);
}

// ---------------- one 16x16 WMMA GEMM pass: C += A[16xK](LDS) * W^T (W is [512 x K], global) ----------------
__device__ inline f32x8 wmma_accum(f32x8 c, const _Float16* __restrict__ Albase, size_t astride,
                                   int K, const _Float16* __restrict__ W,
                                   int tm, int ncol, int lm, int lhi) {
  // A: lane holds 16 contiguous K halves (lanes0-15 K[0..15], lanes16-31 K[16..31])
  const _Float16* ar = Albase + (size_t)(tm * 16 + lm) * astride + lhi * 16;
  // B (same k order as A): lane holds 16 contiguous K halves of weight row ncol (low lanes K0..15, high K16..31)
  const _Float16* br = W + (size_t)ncol * K + lhi * 16;
#pragma unroll 4
  for (int k0 = 0; k0 < K; k0 += 32) {
    f16x8 alo = *(const f16x8*)(ar + k0);
    f16x8 ahi = *(const f16x8*)(ar + k0 + 8);
    f16x16 a = __builtin_shufflevector(alo, ahi, 0, 1, 2, 3, 4, 5, 6, 7,
                                       8, 9, 10, 11, 12, 13, 14, 15);
    f16x16 b = *(const f16x16*)(br + k0);
    c = __builtin_amdgcn_wmma_f32_16x16x32_f16(false, a, false, b, (short)0, c, false, false);
  }
  return c;
}

// ---------------- fused GEMM + DEXAT state update for one 16x16 tile ----------------
__device__ inline void layer_step(const _Float16* __restrict__ a1l, size_t a1s, int K1,
                                  const _Float16* __restrict__ W1,
                                  const _Float16* __restrict__ a2l,   // own old spikes (LDS)
                                  const _Float16* __restrict__ W2,
                                  const float* __restrict__ bi, const float* __restrict__ bh,
                                  float* __restrict__ mem_g, float* __restrict__ b1_g,
                                  float* __restrict__ b2_g, _Float16* __restrict__ s_out,
                                  const float* __restrict__ mask, int t,
                                  int tm, int ncol, int lm, int lhi,
                                  float* hs, float* hm, float* ht) {
  f32x8 c = {0.f, 0.f, 0.f, 0.f, 0.f, 0.f, 0.f, 0.f};
  c = wmma_accum(c, a1l, a1s, K1, W1, tm, ncol, lm, lhi);         // feed-forward term
  c = wmma_accum(c, a2l, HDIM, HDIM, W2, tm, ncol, lm, lhi);      // recurrent term
  float bias = bi[ncol] + bh[ncol];
  float mk = mask[(size_t)ncol * STEPS + t];                      // mask depends on hidden idx only
#pragma unroll
  for (int r = 0; r < 8; ++r) {
    int m = tm * 16 + lhi * 8 + r;                                // C layout: rows r / r+8
    size_t idx = (size_t)m * HDIM + ncol;
    float h = c[r] + bias;
    float so = (float)a2l[idx];                                   // old spike (from LDS stage)
    float b1 = RO1 * b1_g[idx] + (1.f - RO1) * so;
    float b2 = RO2 * b2_g[idx] + (1.f - RO2) * so;
    float Bth = 0.01f + 1.8f * b1 + 9.0f * b2;
    float mem = mem_g[idx];
    float nm = mem * ALPHA + (1.f - ALPHA) * h - Bth * so;
    mem = (mk == 0.f) ? mem : nm;
    float spk = ((mem - Bth) > 0.f ? 1.f : 0.f) * mk;
    mem_g[idx] = mem;
    b1_g[idx] = b1;
    b2_g[idx] = b2;
    s_out[idx] = (_Float16)spk;
    if (hs) {
      size_t ho = ((size_t)m * STEPS + t) * HDIM + ncol;
      hs[ho] = spk; hm[ho] = mem; ht[ho] = Bth;
    }
  }
}

// ---------------- readout projection out[t] = m3 @ w_o.T + b_o (640 dots over wave0s) ----------------
__device__ inline void out_proj(float* __restrict__ out, const float* __restrict__ m3,
                                const float* __restrict__ wo, const float* __restrict__ bo,
                                int t, int blk, int lane) {
  for (int j = lane; j < 40; j += 32) {
    int p = blk * 40 + j;                    // 16 blocks * 40 = 640 = 64*10
    int nb = p / ODIM, o = p % ODIM;
    const float* mr = m3 + (size_t)nb * HDIM;
    const float* wr = wo + (size_t)o * HDIM;
    float acc = bo[o];
#pragma unroll 4
    for (int k = 0; k < HDIM; k += 4) {
      float4 mv = *(const float4*)(mr + k);
      float4 wv = *(const float4*)(wr + k);
      acc = fmaf(mv.x, wv.x, fmaf(mv.y, wv.y, fmaf(mv.z, wv.z, fmaf(mv.w, wv.w, acc))));
    }
    out[((size_t)nb * STEPS + t) * ODIM + o] = acc;
  }
}

// ---------------- init: f16 conversion of x + weights, state init, control reset ----------------
__global__ __launch_bounds__(256) void srnn_init(
    const float* __restrict__ x, const float* __restrict__ wi1, const float* __restrict__ wh1,
    const float* __restrict__ wi2, const float* __restrict__ wh2, const float* __restrict__ wi3,
    const float* __restrict__ wh3, float* __restrict__ out, void* wsv) {
  char* W = (char*)wsv;
  size_t g = (size_t)blockIdx.x * blockDim.x + threadIdx.x;
  size_t gs = (size_t)gridDim.x * blockDim.x;
  _Float16* xh = (_Float16*)(W + W_XH);
  for (size_t i = g; i < XH_E; i += gs) xh[i] = (_Float16)x[i];
  _Float16* w1 = (_Float16*)(W + W_WI1);
  for (size_t i = g; i < (size_t)HDIM * INDIM; i += gs) w1[i] = (_Float16)wi1[i];
  const float* src[5] = {wh1, wi2, wh2, wi3, wh3};
  _Float16* dst[5] = {(_Float16*)(W + W_WH1), (_Float16*)(W + W_WI2), (_Float16*)(W + W_WH2),
                      (_Float16*)(W + W_WI3), (_Float16*)(W + W_WH3)};
  for (int k = 0; k < 5; ++k)
    for (size_t i = g; i < (size_t)HDIM * HDIM; i += gs) dst[k][i] = (_Float16)src[k][i];
  _Float16* sb = (_Float16*)(W + W_SB);
  for (size_t i = g; i < 6 * NH; i += gs) sb[i] = (_Float16)0.f;
  float* bst = (float*)(W + W_BST);
  for (size_t i = g; i < 6 * NH; i += gs) bst[i] = 0.01f;
  float* mst = out + O_M1;
  for (size_t i = g; i < 3 * NH; i += gs) mst[i] = 0.f;
  if (g == 0) { unsigned* c = (unsigned*)(W + W_CTRL); c[0] = 0; c[1] = 0; c[2] = 0; }
}

// ---------------- persistent scan kernel ----------------
__global__ __launch_bounds__(TPB) void srnn_main(
    const float* __restrict__ bi1, const float* __restrict__ bh1,
    const float* __restrict__ bi2, const float* __restrict__ bh2,
    const float* __restrict__ bi3, const float* __restrict__ bh3,
    const float* __restrict__ wo, const float* __restrict__ bo,
    const float* __restrict__ mask1, const float* __restrict__ mask2,
    const float* __restrict__ mask3, float* __restrict__ out, void* wsv) {
  extern __shared__ char smem[];
  _Float16* a1l = (_Float16*)(smem + LDS_A1);
  _Float16* a2l = (_Float16*)(smem + LDS_A2);
  char* W = (char*)wsv;
  unsigned* ctrl = (unsigned*)(W + W_CTRL);
  _Float16* xh = (_Float16*)(W + W_XH);
  const _Float16* wi1h = (const _Float16*)(W + W_WI1);
  const _Float16* wh1h = (const _Float16*)(W + W_WH1);
  const _Float16* wi2h = (const _Float16*)(W + W_WI2);
  const _Float16* wh2h = (const _Float16*)(W + W_WH2);
  const _Float16* wi3h = (const _Float16*)(W + W_WI3);
  const _Float16* wh3h = (const _Float16*)(W + W_WH3);
  _Float16* sb = (_Float16*)(W + W_SB);
  _Float16* s1b[2] = {sb,          sb + NH};
  _Float16* s2b[2] = {sb + 2 * NH, sb + 3 * NH};
  _Float16* s3b[2] = {sb + 4 * NH, sb + 5 * NH};
  float* bst = (float*)(W + W_BST);
  float* b11 = bst;          float* b12 = bst + NH;
  float* b21 = bst + 2 * NH; float* b22 = bst + 3 * NH;
  float* b31 = bst + 4 * NH; float* b32 = bst + 5 * NH;
  float* m1 = out + O_M1; float* m2 = out + O_M2; float* m3 = out + O_M3;
  float* hs = out + O_HS; float* hm = out + O_HM; float* ht = out + O_HT;

  int lane = threadIdx.x & 31, wid = threadIdx.x >> 5;
  int gw = blockIdx.x * 8 + wid;           // 0..127 -> one 16x16 tile of [64 x 512]
  int tm = gw & 3, tn = gw >> 2;
  int lm = lane & 15, lhi = lane >> 4;
  int ncol = tn * 16 + lm;

  for (int t = 0; t < STEPS; ++t) {
    int ri = t & 1, wix = (t + 1) & 1;     // ping-pong read/write indices
    // ---- layer 1: stage x[t] tile (strided 2D) + s1_old via TDM, then GEMM+update ----
    if (wid == 0) {
      tdm_stage(LDS_A1, xh + (size_t)t * INDIM, INDIM, NBATCH, (size_t)STEPS * INDIM);
      tdm_stage(LDS_A2, s1b[ri], HDIM, NBATCH, HDIM);
      __builtin_amdgcn_s_wait_tensorcnt(0);
    }
    __syncthreads();
    layer_step(a1l, INDIM, INDIM, wi1h, a2l, wh1h, bi1, bh1, m1, b11, b12, s1b[wix],
               mask1, t, tm, ncol, lm, lhi, nullptr, nullptr, nullptr);
    if (wid == 0 && t > 0) out_proj(out, m3, wo, bo, t - 1, blockIdx.x, lane);  // overlap readout(t-1)
    grid_barrier(ctrl, ctrl + 1);
    // ---- layer 2 reads s1_new ----
    if (wid == 0) {
      tdm_stage(LDS_A1, s1b[wix], HDIM, NBATCH, HDIM);
      tdm_stage(LDS_A2, s2b[ri], HDIM, NBATCH, HDIM);
      __builtin_amdgcn_s_wait_tensorcnt(0);
    }
    __syncthreads();
    layer_step(a1l, HDIM, HDIM, wi2h, a2l, wh2h, bi2, bh2, m2, b21, b22,
               s2b[wix], mask2, t, tm, ncol, lm, lhi, nullptr, nullptr, nullptr);
    grid_barrier(ctrl, ctrl + 1);
    // ---- layer 3 reads s2_new; also emits hidden spike/mem/theta traces ----
    if (wid == 0) {
      tdm_stage(LDS_A1, s2b[wix], HDIM, NBATCH, HDIM);
      tdm_stage(LDS_A2, s3b[ri], HDIM, NBATCH, HDIM);
      __builtin_amdgcn_s_wait_tensorcnt(0);
    }
    __syncthreads();
    layer_step(a1l, HDIM, HDIM, wi3h, a2l, wh3h, bi3, bh3, m3, b31, b32,
               s3b[wix], mask3, t, tm, ncol, lm, lhi, hs, hm, ht);
    grid_barrier(ctrl, ctrl + 1);
  }
  if (wid == 0) out_proj(out, m3, wo, bo, STEPS - 1, blockIdx.x, lane);

  // final spikes (last write at t=783 landed in buffer index 0) + exact fire count
  size_t gt = (size_t)blockIdx.x * TPB + threadIdx.x;
  unsigned nz = 0;
  for (size_t i = gt; i < 3 * NH; i += (size_t)NWG * TPB) {
    size_t L = i / NH, e = i - L * NH;
    float v = (float)sb[(2 * L) * NH + e];
    out[O_S1 + i] = v;
    if (v != 0.f) nz++;
  }
  if (nz) atomicAdd(ctrl + 2, nz);
  grid_barrier(ctrl, ctrl + 1);
  if (blockIdx.x == 0 && threadIdx.x == 0)
    out[O_FIRE] = (float)__hip_atomic_load(ctrl + 2, __ATOMIC_RELAXED, __HIP_MEMORY_SCOPE_AGENT) /
                  (float)(3 * NH);
}

extern "C" void kernel_launch(void* const* d_in, const int* in_sizes, int n_in,
                              void* d_out, int out_size, void* d_ws, size_t ws_size,
                              hipStream_t stream) {
  const float* x   = (const float*)d_in[0];
  const float* wi1 = (const float*)d_in[1];
  const float* bi1 = (const float*)d_in[2];
  const float* wh1 = (const float*)d_in[3];
  const float* bh1 = (const float*)d_in[4];
  const float* wi2 = (const float*)d_in[5];
  const float* bi2 = (const float*)d_in[6];
  const float* wh2 = (const float*)d_in[7];
  const float* bh2 = (const float*)d_in[8];
  const float* wi3 = (const float*)d_in[9];
  const float* bi3 = (const float*)d_in[10];
  const float* wh3 = (const float*)d_in[11];
  const float* bh3 = (const float*)d_in[12];
  const float* wo  = (const float*)d_in[13];
  const float* bo  = (const float*)d_in[14];
  const float* mk1 = (const float*)d_in[15];
  const float* mk2 = (const float*)d_in[16];
  const float* mk3 = (const float*)d_in[17];
  float* out = (float*)d_out;

  srnn_init<<<2048, 256, 0, stream>>>(x, wi1, wh1, wi2, wh2, wi3, wh3, out, d_ws);
  srnn_main<<<NWG, TPB, LDS_BYTES, stream>>>(bi1, bh1, bi2, bh2, bi3, bh3, wo, bo,
                                             mk1, mk2, mk3, out, d_ws);
}
